// CausalAttention_77893526880318
// MI455X (gfx1250) — matrix operand order repaired and checked by reading, not compiled, not run on hardware
//
#include <hip/hip_runtime.h>
#include <hip/hip_bf16.h>

// Flash attention, causal, fp32 I/O, f16 WMMA compute, fp32 accumulate.
// B=4 H=16 S=2048 D=64. Block = 128 threads = 4 wave32; each wave owns a
// 16-row query sub-tile; block streams 64-key K/V tiles through LDS using
// gfx1250 async global->LDS copies (ASYNCcnt) with double-buffered staging.

typedef __attribute__((ext_vector_type(16))) _Float16 v16h;
typedef __attribute__((ext_vector_type(8)))  float    v8f;

#define B_ 4
#define H_ 16
#define S_ 2048
#define D_ 64
#define BM 64          // query rows per block
#define KT 64          // keys per tile
#define NTHREADS 128   // 4 waves

union FragU { uint4 u[2]; v16h v; };

// A-matrix (16x32 f16) fragment: per ISA 7.12.2, lane L holds M=L%16;
// elements 0-7 -> K = off+0..7, elements 8-15 -> K = 16+off+0..7,
// off = 8 for lanes 16-31. rowBase points at (row, chunkBase) in a
// row-major LDS tile whose K-dim is contiguous.
__device__ __forceinline__ v16h load_frag_a(const _Float16* rowBase, int lane) {
    const int off = (lane & 16) ? 8 : 0;
    FragU f;
    f.u[0] = *(const uint4*)(rowBase + off);
    f.u[1] = *(const uint4*)(rowBase + 16 + off);
    return f.v;
}

// B-matrix (32x16 f16) fragment: same K order as A: elements 0-7 -> K = off+0..7,
// elements 8-15 -> K = 16+off+0..7, off = 8 for lanes 16-31. rowBase points at the
// LDS row for this lane's N column; K-dim contiguous.
__device__ __forceinline__ v16h load_frag_b(const _Float16* rowBase, int lane) {
    const int off = (lane & 16) ? 8 : 0;
    FragU f;
    f.u[0] = *(const uint4*)(rowBase + off);
    f.u[1] = *(const uint4*)(rowBase + 16 + off);
    return f.v;
}

// Row-wise reductions across the 16 lanes of a half-wave (C/D layout puts
// the 16 N columns of a row across lanes 0-15 / 16-31).
__device__ __forceinline__ float redmax16(float v) {
#pragma unroll
    for (int m = 1; m < 16; m <<= 1) v = fmaxf(v, __shfl_xor(v, m, 32));
    return v;
}
__device__ __forceinline__ float redsum16(float v) {
#pragma unroll
    for (int m = 1; m < 16; m <<= 1) v += __shfl_xor(v, m, 32);
    return v;
}

// gfx1250 async copy: 16B global -> LDS, tracked with ASYNCcnt (ISA §10, §15.18).
__device__ __forceinline__ void async_load_b128(unsigned int lds_off, const void* gptr) {
    asm volatile("global_load_async_to_lds_b128 %0, %1, off"
                 :
                 : "v"(lds_off), "v"((unsigned long long)(uintptr_t)gptr)
                 : "memory");
}

#define WMMA(a, b, c) \
    __builtin_amdgcn_wmma_f32_16x16x32_f16(false, (a), false, (b), (short)0, (c), false, false)

__global__ __launch_bounds__(NTHREADS)
void CausalAttention_fa_kernel(const float* __restrict__ q,
                               const float* __restrict__ k,
                               const float* __restrict__ v,
                               float* __restrict__ out) {
    // f32 staging for async copies (double-buffered), f16 compute tiles.
    __shared__ __align__(16) float    Kst[2][KT * D_];   // 2 x 16 KB
    __shared__ __align__(16) float    Vst[2][KT * D_];   // 2 x 16 KB
    __shared__ __align__(16) _Float16 Qs[BM * D_];       // [row][d]    8 KB
    __shared__ __align__(16) _Float16 Kf[KT * D_];       // [key][d]    8 KB
    __shared__ __align__(16) _Float16 Vt[D_ * KT];       // [d][key]    8 KB (transposed)
    __shared__ __align__(16) _Float16 Ps[4 * 16 * KT];   // per-wave P  8 KB

    const int tid   = threadIdx.x;
    const int lane  = tid & 31;
    const int wave  = tid >> 5;
    const int nCol  = lane & 15;          // N column / M row within frag
    const int rowHi = (lane >> 4) << 3;   // C/D row offset: 0 or 8

    const int    qBase = blockIdx.x * BM;
    const int    bh    = blockIdx.y;
    const size_t base  = (size_t)bh * S_ * D_;
    const float* qg = q + base + (size_t)qBase * D_;
    const float* kg = k + base;
    const float* vg = v + base;
    float*       og = out + base;

    const unsigned int ldsK0 = (unsigned int)(uintptr_t)&Kst[0][0];
    const unsigned int ldsK1 = (unsigned int)(uintptr_t)&Kst[1][0];
    const unsigned int ldsV0 = (unsigned int)(uintptr_t)&Vst[0][0];
    const unsigned int ldsV1 = (unsigned int)(uintptr_t)&Vst[1][0];

    const int numTiles = blockIdx.x + 1;   // causal: only tiles with keys <= qBase+63

    // ---- kick async copy of tile 0 (16 x b128 per thread) ----
    {
        const char* ks = (const char*)kg;
        const char* vs = (const char*)vg;
#pragma unroll
        for (int i = 0; i < 8; ++i) {
            unsigned int boff = (unsigned int)(tid + i * NTHREADS) * 16u;
            async_load_b128(ldsK0 + boff, ks + boff);
            async_load_b128(ldsV0 + boff, vs + boff);
        }
    }

    // ---- Q tile -> LDS as f16, pre-scaled by 1/sqrt(D) (overlaps async) ----
    const float qscale = 0.125f;
#pragma unroll
    for (int i = 0; i < 8; ++i) {
        int off = (tid + i * NTHREADS) * 4;
        float4 qv = *(const float4*)(qg + off);
        union { _Float16 h[4]; uint2 u; } p;
        p.h[0] = (_Float16)(qv.x * qscale);
        p.h[1] = (_Float16)(qv.y * qscale);
        p.h[2] = (_Float16)(qv.z * qscale);
        p.h[3] = (_Float16)(qv.w * qscale);
        *(uint2*)(Qs + off) = p.u;
    }
    __syncthreads();

    // Per-wave Q A-fragments (rows wave*16 .. wave*16+15), d-chunks 0..31 / 32..63.
    const v16h aQ0 = load_frag_a(Qs + (wave * 16 + nCol) * D_ + 0,  lane);
    const v16h aQ1 = load_frag_a(Qs + (wave * 16 + nCol) * D_ + 32, lane);

    v8f acc[4] = {};
    float mRun[8], lRun[8];
#pragma unroll
    for (int r = 0; r < 8; ++r) { mRun[r] = -__builtin_inff(); lRun[r] = 0.f; }

    const int qRow0 = qBase + wave * 16;   // wave's first query row
    _Float16* Pw = Ps + wave * 16 * KT;

    for (int t = 0; t < numTiles; ++t) {
        const int kBase = t * KT;
        const int buf   = t & 1;

        // ---- issue async copy of tile t+1 into the other staging buffer ----
        if (t + 1 < numTiles) {
            const char* ks = (const char*)(kg + (size_t)(kBase + KT) * D_);
            const char* vs = (const char*)(vg + (size_t)(kBase + KT) * D_);
            const unsigned int ldsK = buf ? ldsK0 : ldsK1;
            const unsigned int ldsV = buf ? ldsV0 : ldsV1;
#pragma unroll
            for (int i = 0; i < 8; ++i) {
                unsigned int boff = (unsigned int)(tid + i * NTHREADS) * 16u;
                async_load_b128(ldsK + boff, ks + boff);
                async_load_b128(ldsV + boff, vs + boff);
            }
            // 16 newest ops belong to tile t+1 -> tile t's copies have landed.
            asm volatile("s_wait_asynccnt 0x10" ::: "memory");
        } else {
            asm volatile("s_wait_asynccnt 0x0" ::: "memory");
        }
        __syncthreads();   // tile t staged for all waves

        // ---- convert stage f32 -> Kf ([key][d] f16) and Vt ([d][key] f16) ----
        {
            const float* ks = Kst[buf];
            const float* vs = Vst[buf];
#pragma unroll
            for (int i = 0; i < 8; ++i) {
                int off = (tid + i * NTHREADS) * 4;
                float4 kv = *(const float4*)(ks + off);
                union { _Float16 h[4]; uint2 u; } p;
                p.h[0] = (_Float16)kv.x; p.h[1] = (_Float16)kv.y;
                p.h[2] = (_Float16)kv.z; p.h[3] = (_Float16)kv.w;
                *(uint2*)(Kf + off) = p.u;

                float4 vv = *(const float4*)(vs + off);
                int key = off >> 6;    // off / D_
                int d0  = off & 63;    // off % D_
                Vt[(d0 + 0) * KT + key] = (_Float16)vv.x;
                Vt[(d0 + 1) * KT + key] = (_Float16)vv.y;
                Vt[(d0 + 2) * KT + key] = (_Float16)vv.z;
                Vt[(d0 + 3) * KT + key] = (_Float16)vv.w;
            }
        }
        __syncthreads();   // f16 tiles ready

        // Wave-uniform causal skip: tile entirely above this wave's rows.
        if (kBase <= qRow0 + 15) {
            // ---- scores S = (Q/sqrt(D)) K^T, four 16-key sub-tiles ----
            v8f s[4] = {};
#pragma unroll
            for (int nt = 0; nt < 4; ++nt) {
                v16h b0 = load_frag_b(Kf + (nt * 16 + nCol) * D_ + 0,  lane);
                s[nt] = WMMA(aQ0, b0, s[nt]);
                v16h b1 = load_frag_b(Kf + (nt * 16 + nCol) * D_ + 32, lane);
                s[nt] = WMMA(aQ1, b1, s[nt]);
            }

            // ---- causal mask (boundary tile only) ----
            if (kBase + KT - 1 > qRow0) {
#pragma unroll
                for (int r = 0; r < 8; ++r) {
                    int qr = qRow0 + r + rowHi;
#pragma unroll
                    for (int nt = 0; nt < 4; ++nt)
                        if (kBase + nt * 16 + nCol > qr) s[nt][r] = -__builtin_inff();
                }
            }

            // ---- online softmax update ----
            float alpha[8];
#pragma unroll
            for (int r = 0; r < 8; ++r) {
                float mx = fmaxf(fmaxf(s[0][r], s[1][r]), fmaxf(s[2][r], s[3][r]));
                mx = redmax16(mx);
                float mNew = fmaxf(mRun[r], mx);
                alpha[r]   = __expf(mRun[r] - mNew);
                mRun[r]    = mNew;
                float p0 = __expf(s[0][r] - mNew);
                float p1 = __expf(s[1][r] - mNew);
                float p2 = __expf(s[2][r] - mNew);
                float p3 = __expf(s[3][r] - mNew);
                s[0][r] = p0; s[1][r] = p1; s[2][r] = p2; s[3][r] = p3;
                lRun[r] = lRun[r] * alpha[r] + redsum16((p0 + p1) + (p2 + p3));
            }
#pragma unroll
            for (int j = 0; j < 4; ++j)
#pragma unroll
                for (int r = 0; r < 8; ++r) acc[j][r] *= alpha[r];

            // ---- D-layout P -> LDS -> A-layout fragments ----
#pragma unroll
            for (int r = 0; r < 8; ++r)
#pragma unroll
                for (int nt = 0; nt < 4; ++nt)
                    Pw[(r + rowHi) * KT + nt * 16 + nCol] = (_Float16)s[nt][r];
            asm volatile("s_wait_dscnt 0x0" ::: "memory");  // wave-private RAW

            const v16h aP0 = load_frag_a(Pw + nCol * KT + 0,  lane);
            const v16h aP1 = load_frag_a(Pw + nCol * KT + 32, lane);
#pragma unroll
            for (int j = 0; j < 4; ++j) {
                v16h b0 = load_frag_b(Vt + (j * 16 + nCol) * KT + 0,  lane);
                acc[j] = WMMA(aP0, b0, acc[j]);
                v16h b1 = load_frag_b(Vt + (j * 16 + nCol) * KT + 32, lane);
                acc[j] = WMMA(aP1, b1, acc[j]);
            }
        }
    }

    // ---- epilogue: O = acc / l ----
#pragma unroll
    for (int r = 0; r < 8; ++r) {
        float  inv  = 1.0f / lRun[r];
        int    qr   = qRow0 + r + rowHi;
        float* orow = og + (size_t)qr * D_;
#pragma unroll
        for (int j = 0; j < 4; ++j)
            orow[j * 16 + nCol] = acc[j][r] * inv;
    }
}

extern "C" void kernel_launch(void* const* d_in, const int* in_sizes, int n_in,
                              void* d_out, int out_size, void* d_ws, size_t ws_size,
                              hipStream_t stream) {
    const float* q = (const float*)d_in[0];
    const float* k = (const float*)d_in[1];
    const float* v = (const float*)d_in[2];
    float* out = (float*)d_out;
    (void)in_sizes; (void)n_in; (void)out_size; (void)d_ws; (void)ws_size;

    dim3 grid(S_ / BM, B_ * H_);   // (32, 64)
    dim3 block(NTHREADS);
    CausalAttention_fa_kernel<<<grid, block, 0, stream>>>(q, k, v, out);
}
